// ONNXMultiheadAttention_59485297050051
// MI455X (gfx1250) — matrix operand order repaired and checked by reading, not compiled, not run on hardware
//
#include <hip/hip_runtime.h>
#include <stdint.h>

#define S_LEN  2048
#define BATCH  4
#define DMODEL 1024
#define NHEADS 16
#define HDIM   64
#define MROWS  (S_LEN * BATCH) /* 8192 */

typedef __bf16 v16bf __attribute__((ext_vector_type(16)));
typedef float  v8f   __attribute__((ext_vector_type(8)));

// Branch-free f32 -> bf16 via the hardware conversion (gfx1250 has native bf16 cvt).
__device__ __forceinline__ unsigned short f2bf(float f) {
  return __builtin_bit_cast(unsigned short, static_cast<__bf16>(f));
}
__device__ __forceinline__ __bf16 bfbits(unsigned short u) {
  return __builtin_bit_cast(__bf16, u);
}
// A-matrix (16-bit, 16x32) element->K map: contiguous, K = e + 16*half (matches B fills)
__device__ __forceinline__ int kmapA(int e, int half) {
  return e + 16 * half;
}

// gfx1250 async global->LDS copy (ASYNCcnt-tracked). lds_off = low 32 bits of a
// generic pointer into __shared__ (LDS aperture keeps the offset in addr[31:0]).
__device__ __forceinline__ void async_g2l_b128(unsigned lds_off, const void* gptr) {
  asm volatile("global_load_async_to_lds_b128 %0, %1, off"
               :: "v"(lds_off), "v"(gptr) : "memory");
}
__device__ __forceinline__ void wait_async0() {
  asm volatile("s_wait_asynccnt 0x0" ::: "memory");
}

// ---------------------------------------------------------------------------
// GEMM: Y[m,n] = sum_k X[m,k] * W[n,k] + bias[n]   (torch Linear convention)
// M=8192, N=K=1024. Block tile 128x64, 8 waves; wave w -> rows w*16..+15, all
// 64 cols (4 WMMA n-tiles). mode 0: fp32 row-major out. mode 1: bf16 out in
// attention layout (B,H,S,HD) with m = s*B+b, n = h*HD+hd.
// ---------------------------------------------------------------------------
__global__ __launch_bounds__(256)
void gemm_xwt_wmma(const float* __restrict__ X, const float* __restrict__ W,
                   const float* __restrict__ bias, float* __restrict__ outF,
                   unsigned short* __restrict__ outB, int mode)
{
  __shared__ unsigned short As[128 * 36]; // padded stride 36 shorts (72B)
  __shared__ unsigned short Bs[64 * 36];

  const int tid  = threadIdx.x;
  const int lane = tid & 31, w = tid >> 5;
  const int half = lane >> 4, lm = lane & 15;
  const int m0 = blockIdx.x * 128;
  const int n0 = blockIdx.y * 64;

  v8f acc[4] = {};

  for (int k0 = 0; k0 < DMODEL; k0 += 32) {
    // prefetch next K-step (one 128B line per row)
    if (k0 + 32 < DMODEL) {
      int r = tid >> 1;
      __builtin_prefetch(X + (size_t)(m0 + r) * DMODEL + k0 + 32, 0, 1);
      if (tid < 64) __builtin_prefetch(W + (size_t)(n0 + tid) * DMODEL + k0 + 32, 0, 1);
    }
    // stage A tile 128x32 (fp32 -> bf16), float4 per thread x4
#pragma unroll
    for (int c = 0; c < 4; ++c) {
      int idx = tid + c * 256;
      int r = idx >> 3, kc = (idx & 7) * 4;
      const float4 v = *reinterpret_cast<const float4*>(X + (size_t)(m0 + r) * DMODEL + k0 + kc);
      As[r * 36 + kc + 0] = f2bf(v.x);
      As[r * 36 + kc + 1] = f2bf(v.y);
      As[r * 36 + kc + 2] = f2bf(v.z);
      As[r * 36 + kc + 3] = f2bf(v.w);
    }
    // stage W tile 64x32: rows of W contiguous in k -> coalesced
#pragma unroll
    for (int c = 0; c < 2; ++c) {
      int idx = tid + c * 256;
      int r = idx >> 3, kc = (idx & 7) * 4;
      const float4 v = *reinterpret_cast<const float4*>(W + (size_t)(n0 + r) * DMODEL + k0 + kc);
      Bs[r * 36 + kc + 0] = f2bf(v.x);
      Bs[r * 36 + kc + 1] = f2bf(v.y);
      Bs[r * 36 + kc + 2] = f2bf(v.z);
      Bs[r * 36 + kc + 3] = f2bf(v.w);
    }
    __syncthreads();

    // Load A fragment and ALL four B fragments first, then issue 4 WMMAs
    // back-to-back (one dscnt wait instead of four).
    v16bf a;
    const unsigned short* arow = &As[(w * 16 + lm) * 36];
#pragma unroll
    for (int e = 0; e < 16; ++e) a[e] = bfbits(arow[kmapA(e, half)]);

    v16bf b[4];
#pragma unroll
    for (int nt = 0; nt < 4; ++nt) {
      const unsigned short* brow = &Bs[(nt * 16 + lm) * 36]; // lane = output col n
#pragma unroll
      for (int e = 0; e < 16; ++e) b[nt][e] = bfbits(brow[e + 16 * half]); // elem e -> K=e+16*half
    }
#pragma unroll
    for (int nt = 0; nt < 4; ++nt)
      acc[nt] = __builtin_amdgcn_wmma_f32_16x16x32_bf16(false, a, false, b[nt],
                                                        (short)0, acc[nt], false, false);
    __syncthreads();
  }

  // epilogue: C layout -> row = e + 8*half, col = lane&15
#pragma unroll
  for (int nt = 0; nt < 4; ++nt) {
#pragma unroll
    for (int e = 0; e < 8; ++e) {
      int row = w * 16 + e + 8 * half;
      int col = nt * 16 + lm;
      int m = m0 + row, n = n0 + col;
      float v = acc[nt][e] + bias[n];
      if (mode == 0) {
        outF[(size_t)m * DMODEL + n] = v;
      } else {
        int s = m / BATCH, bb = m % BATCH;
        int h = n / HDIM, hd = n % HDIM;
        outB[(((size_t)bb * NHEADS + h) * S_LEN + s) * HDIM + hd] = f2bf(v);
      }
    }
  }
}

// ---------------------------------------------------------------------------
// Flash attention: grid (S/128, B*H). Wave w owns 16 q-rows; Q fragments stay
// in registers across the whole KV sweep. K tile staged with async
// global->LDS b128 copies (ASYNCcnt path); V staged transposed manually.
// Online softmax in fp32.
// ---------------------------------------------------------------------------
__global__ __launch_bounds__(256)
void flash_attn_wmma(const unsigned short* __restrict__ Qb,
                     const unsigned short* __restrict__ Kb,
                     const unsigned short* __restrict__ Vb,
                     const float* __restrict__ attn_mask,
                     const unsigned char* __restrict__ kpm,
                     float* __restrict__ O)
{
  __shared__ unsigned short Kt[64 * 72];      // [kv][hd], 144B rows: 16B aligned, 36-bank stride
  __shared__ unsigned short Vt[64 * 68];      // [hd][kv] (transposed), padded
  __shared__ unsigned short Pb[8 * 16 * 68];  // per-wave P tile [16][64], padded

  const int tid  = threadIdx.x;
  const int lane = tid & 31, w = tid >> 5;
  const int half = lane >> 4, lm = lane & 15;
  const int q0 = blockIdx.x * 128;
  const int bh = blockIdx.y;
  const int bb = bh / NHEADS, h = bh % NHEADS;

  const unsigned ktbase = (unsigned)(uintptr_t)&Kt[0]; // LDS byte offset

  // Q fragments: two K-steps of 32 covering HD=64
  const int qi = q0 + w * 16 + lm;
  const unsigned short* qrow = Qb + ((size_t)bh * S_LEN + qi) * HDIM;
  v16bf qa0, qa1;
#pragma unroll
  for (int e = 0; e < 16; ++e) {
    int kk = kmapA(e, half);
    qa0[e] = bfbits(qrow[kk]);
    qa1[e] = bfbits(qrow[kk + 32]);
  }

  float mrow[8], lrow[8];
  v8f acc[4] = {};
#pragma unroll
  for (int e = 0; e < 8; ++e) { mrow[e] = -3.0e38f; lrow[e] = 0.f; }
  const float scale = 0.125f; // 1/sqrt(64)

  for (int kv0 = 0; kv0 < S_LEN; kv0 += 64) {
    // --- stage K tile via async global->LDS (64 rows x 128B = 512 b128 xfers)
    {
      const char* kg = (const char*)(Kb + ((size_t)bh * S_LEN + kv0) * HDIM);
#pragma unroll
      for (int c = 0; c < 2; ++c) {
        int i = tid + c * 256;             // 0..511
        int row = i >> 3, seg = i & 7;
        async_g2l_b128(ktbase + row * 144 + seg * 16, kg + row * 128 + seg * 16);
      }
    }
    // --- stage V transposed (manual: scatter b16)
    {
      int j = tid >> 2, seg = (tid & 3) * 16;
      const unsigned short* vrow = Vb + ((size_t)bh * S_LEN + kv0 + j) * HDIM + seg;
#pragma unroll
      for (int i = 0; i < 16; ++i) Vt[(seg + i) * 68 + j] = vrow[i];
    }
    wait_async0();
    __syncthreads();

    // scores: S = Q (16x64) @ K^T (64 x 64kv) -> 4 n-tiles, 2 k-steps each
    v8f sc[4];
#pragma unroll
    for (int nt = 0; nt < 4; ++nt) {
      v16bf b0, b1;
      const unsigned short* kr = &Kt[(nt * 16 + lm) * 72]; // lane = kv col
#pragma unroll
      for (int e = 0; e < 16; ++e) {
        int kk = e + 16 * half;
        b0[e] = bfbits(kr[kk]);
        b1[e] = bfbits(kr[kk + 32]);
      }
      v8f c = {};
      c = __builtin_amdgcn_wmma_f32_16x16x32_bf16(false, qa0, false, b0, (short)0, c, false, false);
      c = __builtin_amdgcn_wmma_f32_16x16x32_bf16(false, qa1, false, b1, (short)0, c, false, false);
      sc[nt] = c;
    }

    // scale + attn_mask + key_padding_mask (C layout addressing)
#pragma unroll
    for (int nt = 0; nt < 4; ++nt) {
      int j = kv0 + nt * 16 + lm;
      bool pad = kpm[(size_t)bb * S_LEN + j] != 0;
#pragma unroll
      for (int e = 0; e < 8; ++e) {
        int q = q0 + w * 16 + e + 8 * half;
        float s = sc[nt][e] * scale + attn_mask[(size_t)q * S_LEN + j];
        sc[nt][e] = pad ? -3.0e38f : s;
      }
    }

    // online softmax: element e = one q-row, spread across 16 lanes of a half
#pragma unroll
    for (int e = 0; e < 8; ++e) {
      float mx = fmaxf(fmaxf(sc[0][e], sc[1][e]), fmaxf(sc[2][e], sc[3][e]));
#pragma unroll
      for (int msk = 1; msk < 16; msk <<= 1) mx = fmaxf(mx, __shfl_xor(mx, msk, 32));
      float mnew = fmaxf(fmaxf(mrow[e], mx), -3.0e38f);
      float corr = __expf(mrow[e] - mnew);
      float lsum = 0.f;
#pragma unroll
      for (int nt = 0; nt < 4; ++nt) {
        float p = __expf(sc[nt][e] - mnew);
        lsum += p;
        Pb[(w * 16 + e + 8 * half) * 68 + nt * 16 + lm] = f2bf(p); // C-layout -> row-major LDS
      }
#pragma unroll
      for (int msk = 1; msk < 16; msk <<= 1) lsum += __shfl_xor(lsum, msk, 32);
      lrow[e] = lrow[e] * corr + lsum;
      mrow[e] = mnew;
#pragma unroll
      for (int ot = 0; ot < 4; ++ot) acc[ot][e] *= corr;
    }

    // O += P (16 x 64kv) @ V (64kv x 64hd): 2 k-steps, 4 hd-tiles
#pragma unroll
    for (int ks = 0; ks < 2; ++ks) {
      v16bf ap;
      const unsigned short* prow = &Pb[(w * 16 + lm) * 68 + 32 * ks];
#pragma unroll
      for (int e = 0; e < 16; ++e) ap[e] = bfbits(prow[kmapA(e, half)]);
      v16bf bv[4];
#pragma unroll
      for (int ot = 0; ot < 4; ++ot) {
        const unsigned short* vr = &Vt[(ot * 16 + lm) * 68 + 32 * ks]; // lane = hd col
#pragma unroll
        for (int e = 0; e < 16; ++e) bv[ot][e] = bfbits(vr[e + 16 * half]);
      }
#pragma unroll
      for (int ot = 0; ot < 4; ++ot)
        acc[ot] = __builtin_amdgcn_wmma_f32_16x16x32_bf16(false, ap, false, bv[ot],
                                                          (short)0, acc[ot], false, false);
    }
    __syncthreads();
  }

  // write O in (S,B,D) fp32, normalized by row sum
#pragma unroll
  for (int ot = 0; ot < 4; ++ot) {
#pragma unroll
    for (int e = 0; e < 8; ++e) {
      int q  = q0 + w * 16 + e + 8 * half;
      int hd = ot * 16 + lm;
      O[((size_t)q * BATCH + bb) * DMODEL + h * HDIM + hd] = acc[ot][e] * (1.0f / lrow[e]);
    }
  }
}

// ---------------------------------------------------------------------------
extern "C" void kernel_launch(void* const* d_in, const int* in_sizes, int n_in,
                              void* d_out, int out_size, void* d_ws, size_t ws_size,
                              hipStream_t stream)
{
  const float* query = (const float*)d_in[0];
  const float* key   = (const float*)d_in[1];
  const float* value = (const float*)d_in[2];
  const float* attn_mask   = (const float*)d_in[3];
  const unsigned char* kpm = (const unsigned char*)d_in[4];
  const float* Wq = (const float*)d_in[5];
  const float* bq = (const float*)d_in[6];
  const float* Wk = (const float*)d_in[7];
  const float* bk = (const float*)d_in[8];
  const float* Wv = (const float*)d_in[9];
  const float* bv = (const float*)d_in[10];
  const float* Wo = (const float*)d_in[11];
  const float* bo = (const float*)d_in[12];
  float* out = (float*)d_out;

  // workspace: Q,K,V bf16 (B,H,S,HD) + O fp32 (S,B,D)  ~= 84 MB
  const size_t qkvElems = (size_t)BATCH * NHEADS * S_LEN * HDIM;
  unsigned short* Qb = (unsigned short*)d_ws;
  unsigned short* Kb = Qb + qkvElems;
  unsigned short* Vb = Kb + qkvElems;
  float* Obuf = (float*)(Vb + qkvElems);

  dim3 gGemm(MROWS / 128, DMODEL / 64);
  gemm_xwt_wmma<<<gGemm, 256, 0, stream>>>(query, Wq, bq, nullptr, Qb, 1);
  gemm_xwt_wmma<<<gGemm, 256, 0, stream>>>(key,   Wk, bk, nullptr, Kb, 1);
  gemm_xwt_wmma<<<gGemm, 256, 0, stream>>>(value, Wv, bv, nullptr, Vb, 1);

  dim3 gAttn(S_LEN / 128, BATCH * NHEADS);
  flash_attn_wmma<<<gAttn, 256, 0, stream>>>(Qb, Kb, Vb, attn_mask, kpm, Obuf);

  gemm_xwt_wmma<<<gGemm, 256, 0, stream>>>(Obuf, Wo, bo, out, nullptr, 0);
}
